// GraphAttention_75436805587297
// MI455X (gfx1250) — matrix operand order repaired and checked by reading, not compiled, not run on hardware
//
#include <hip/hip_runtime.h>

#define HH   4
#define BB   8
#define NN   1024
#define FIN  128
#define FOUT 32
#define SLOPE 0.01f

#define KP   512              // k-panel staged in LDS per phase
#define LST  (KP + 8)         // padded LDS row stride (f16 elems): 1040B % 256B = 16B -> conflict-free

typedef __attribute__((ext_vector_type(16))) _Float16 v16h;
typedef __attribute__((ext_vector_type(8)))  _Float16 v8h;
typedef __attribute__((ext_vector_type(8)))  float    v8f;
typedef __attribute__((ext_vector_type(4)))  float    v4f;
typedef __attribute__((ext_vector_type(4)))  int      v4i;

typedef __attribute__((address_space(1))) v4i* gptr4;
typedef __attribute__((address_space(3))) v4i* lptr4;

__device__ __forceinline__ float leaky(float z) { return fmaxf(z, SLOPE * z); }

#if __has_builtin(__builtin_amdgcn_global_load_async_to_lds_b128)
#define USE_ASYNC 1
#else
#define USE_ASYNC 0
#endif

__device__ __forceinline__ void copy16_g2l(const _Float16* g, _Float16* l) {
#if USE_ASYNC
  __builtin_amdgcn_global_load_async_to_lds_b128((gptr4)g, (lptr4)l, 0, 0);
#else
  *(v8h*)l = *(const v8h*)g;
#endif
}

__device__ __forceinline__ void wait_async0() {
#if USE_ASYNC
#if __has_builtin(__builtin_amdgcn_s_wait_asynccnt)
  __builtin_amdgcn_s_wait_asynccnt(0);
#else
  asm volatile("s_wait_asynccnt 0x0" ::: "memory");
#endif
#endif
}

// ---------------------------------------------------------------------------
// Kernel 0: W [H][FIN][FOUT] f32  ->  Wt [H][FOUT][FIN] f16  (k-contiguous)
// ---------------------------------------------------------------------------
__global__ void k_wt(const float* __restrict__ W, _Float16* __restrict__ Wt) {
  int tid = blockIdx.x * blockDim.x + threadIdx.x;   // H*FOUT*FIN = 16384
  int k = tid & (FIN - 1);
  int o = (tid >> 7) & (FOUT - 1);
  int h = tid >> 12;
  Wt[tid] = (_Float16)W[(h * FIN + k) * FOUT + o];
}

// ---------------------------------------------------------------------------
// Kernel 1: projection  h = x * W  via v_wmma_f32_16x16x32_f16.
// One wave: 16-row (n) tile x 32 cols (o) for one (h,b). Writes hT[hb][o][n] f16.
// ---------------------------------------------------------------------------
__global__ void k_proj(const float* __restrict__ x, const _Float16* __restrict__ Wt,
                       _Float16* __restrict__ hT) {
  const int lane = threadIdx.x & 31;
  const int wid  = threadIdx.x >> 5;
  const int gw   = blockIdx.x * (blockDim.x >> 5) + wid;  // 0..2047
  const int hb   = gw >> 6;
  const int n0   = (gw & 63) << 4;
  const int h    = hb >> 3;
  const int bi   = hb & 7;
  const int nl   = lane & 15;
  const int hi   = lane >> 4;
  const int offA = hi << 3;    // unused: A also loads at offB (0 or 16)
  const int offB = hi << 4;    // B k-offset in chunk: 0 or 16

  const float*    xrow = x  + (size_t)(bi * NN + n0 + nl) * FIN;
  const _Float16* wt_h = Wt + (size_t)h * FOUT * FIN;

  v8f c0 = {}; v8f c1 = {};
#pragma unroll
  for (int c = 0; c < FIN; c += 32) {
    v4f g0 = *(const v4f*)(xrow + c + offB);
    v4f g1 = *(const v4f*)(xrow + c + offB + 4);
    v4f g2 = *(const v4f*)(xrow + c + offB + 8);
    v4f g3 = *(const v4f*)(xrow + c + offB + 12);
    v16h av;
#pragma unroll
    for (int i = 0; i < 4; i++) {
      av[i]      = (_Float16)g0[i];
      av[4 + i]  = (_Float16)g1[i];
      av[8 + i]  = (_Float16)g2[i];
      av[12 + i] = (_Float16)g3[i];
    }
    v16h b0 = *(const v16h*)(wt_h + (0 * 16 + nl) * FIN + c + offB);
    v16h b1 = *(const v16h*)(wt_h + (1 * 16 + nl) * FIN + c + offB);
    c0 = __builtin_amdgcn_wmma_f32_16x16x32_f16(false, av, false, b0, (short)0, c0, false, false);
    c1 = __builtin_amdgcn_wmma_f32_16x16x32_f16(false, av, false, b1, (short)0, c1, false, false);
  }
  const int nst = n0 + (hi << 3);
  v8h s0, s1;
#pragma unroll
  for (int v = 0; v < 8; v++) { s0[v] = (_Float16)c0[v]; s1[v] = (_Float16)c1[v]; }
  *(v8h*)(hT + (size_t)(hb * FOUT + nl) * NN + nst)      = s0;
  *(v8h*)(hT + (size_t)(hb * FOUT + 16 + nl) * NN + nst) = s1;
}

// ---------------------------------------------------------------------------
// Kernel 2: s[n] = h[n,:].a_src, d[n] = h[n,:].a_dst, dmax = max_n d[n]
// ---------------------------------------------------------------------------
__global__ void k_scores(const _Float16* __restrict__ hT, const float* __restrict__ a,
                         float* __restrict__ s, float* __restrict__ d,
                         float* __restrict__ dmax) {
  __shared__ float red[NN];
  const int hb = blockIdx.x;
  const int h  = hb >> 3;
  const int n  = threadIdx.x;
  const float* av = a + h * 2 * FOUT;   // uniform -> scalar loads
  float sv = 0.f, dv = 0.f;
#pragma unroll
  for (int o = 0; o < FOUT; o++) {
    float hv = (float)hT[(size_t)(hb * FOUT + o) * NN + n];
    sv += hv * av[o];
    dv += hv * av[FOUT + o];
  }
  s[hb * NN + n] = sv;
  d[hb * NN + n] = dv;
  red[n] = dv;
  __syncthreads();
  for (int t = 512; t > 0; t >>= 1) {
    if (n < t) red[n] = fmaxf(red[n], red[n + t]);
    __syncthreads();
  }
  if (n == 0) dmax[hb] = red[0];
}

// ---------------------------------------------------------------------------
// Kernel 3: fused softmax + att*h. Block = 8 waves sharing one (h,b);
// hT panel staged to LDS via async-to-LDS; WMMA f16->f32 with LDS B operand.
// ---------------------------------------------------------------------------
__global__ void k_attn(const _Float16* __restrict__ hT, const float* __restrict__ s,
                       const float* __restrict__ d, const float* __restrict__ dmax,
                       float* __restrict__ out) {
  __shared__ _Float16 bufh[FOUT * LST];   // 32 x (512+8) f16 ~= 32.5 KB
  __shared__ float sps[8][32];

  const int lane = threadIdx.x & 31;
  const int wid  = threadIdx.x >> 5;
  const int gw   = blockIdx.x * 8 + wid;  // 8 waves/block, same hb for all
  const int hb   = gw >> 6;
  const int n0   = (gw & 63) << 4;
  const int h    = hb >> 3;
  const int bi   = hb & 7;
  const int nl   = lane & 15;
  const int hi   = lane >> 4;
  const int offA = hi << 3;
  const int offB = hi << 4;

  const float sL = s[hb * NN + n0 + nl];
  const float mL = leaky(sL + dmax[hb]);          // exact row max of e
  const float* dp = d + hb * NN;
  const _Float16* hsrc = hT + (size_t)hb * FOUT * NN;

  v8f c0 = {}; v8f c1 = {};
  float rsum = 0.f;

  for (int p = 0; p < NN; p += KP) {
    __syncthreads();   // previous panel fully consumed
    // cooperative stage: FOUT x KP panel (2048 x 16B chunks, 8 per thread)
#pragma unroll
    for (int it = 0; it < (FOUT * KP / 8) / 256; it++) {
      int i    = it * 256 + threadIdx.x;
      int r    = i >> 6;            // KP/8 = 64 chunks per row
      int ccol = (i & 63) << 3;
      copy16_g2l(hsrc + (size_t)r * NN + p + ccol, bufh + r * LST + ccol);
    }
    wait_async0();
    __syncthreads();

    for (int c = 0; c < KP; c += 32) {
      const int kc = p + c;
      v4f q0 = *(const v4f*)(dp + kc + offB);
      v4f q1 = *(const v4f*)(dp + kc + offB + 4);
      v4f q2 = *(const v4f*)(dp + kc + offB + 8);
      v4f q3 = *(const v4f*)(dp + kc + offB + 12);
      if (kc + 32 < NN) __builtin_prefetch(dp + kc + 32, 0, 1);
      float pv[16];
#pragma unroll
      for (int i = 0; i < 4; i++) {
        pv[i] = q0[i]; pv[4 + i] = q1[i]; pv[8 + i] = q2[i]; pv[12 + i] = q3[i];
      }
      v16h pa;
#pragma unroll
      for (int i = 0; i < 16; i++) {
        float e = __expf(leaky(sL + pv[i]) - mL);   // <= 1, no overflow
        rsum += e;
        pa[i] = (_Float16)e;
      }
      // B tiles from LDS (16B-aligned halves; padded stride -> conflict-free)
      const _Float16* r0 = bufh + nl * LST + c + offB;
      const _Float16* r1 = bufh + (16 + nl) * LST + c + offB;
      union { v16h v; v8h half[2]; } u0, u1;
      u0.half[0] = *(const v8h*)r0;  u0.half[1] = *(const v8h*)(r0 + 8);
      u1.half[0] = *(const v8h*)r1;  u1.half[1] = *(const v8h*)(r1 + 8);
      c0 = __builtin_amdgcn_wmma_f32_16x16x32_f16(false, pa, false, u0.v, (short)0, c0, false, false);
      c1 = __builtin_amdgcn_wmma_f32_16x16x32_f16(false, pa, false, u1.v, (short)0, c1, false, false);
    }
  }

  // Row-sum exchange: lane l holds partial for row (l&15); total = part[r]+part[r+16]
  sps[wid][lane] = rsum;
  const int roff = hi << 3;
  v4f a0 = *(const v4f*)&sps[wid][roff];
  v4f a1 = *(const v4f*)&sps[wid][roff + 4];
  v4f e0 = *(const v4f*)&sps[wid][roff + 16];
  v4f e1 = *(const v4f*)&sps[wid][roff + 20];

  const int nst = n0 + roff;
  float* op = out + (size_t)(bi * NN + nst) * (HH * FOUT) + h * FOUT + nl;
#pragma unroll
  for (int v = 0; v < 8; v++) {
    float tot = (v < 4) ? (a0[v] + e0[v]) : (a1[v - 4] + e1[v - 4]);
    float inv = 1.0f / tot;
    op[(size_t)v * (HH * FOUT)]      = c0[v] * inv;   // o = nl
    op[(size_t)v * (HH * FOUT) + 16] = c1[v] * inv;   // o = 16 + nl
  }
}

// ---------------------------------------------------------------------------
extern "C" void kernel_launch(void* const* d_in, const int* in_sizes, int n_in,
                              void* d_out, int out_size, void* d_ws, size_t ws_size,
                              hipStream_t stream) {
  const float* x = (const float*)d_in[0];
  // d_in[1] = adj : unused by the reference math
  const float* W = (const float*)d_in[2];
  const float* a = (const float*)d_in[3];
  float* out = (float*)d_out;

  char* ws = (char*)d_ws;
  size_t off = 0;
  _Float16* hT = (_Float16*)(ws + off); off += (size_t)HH * BB * FOUT * NN * sizeof(_Float16); // 2 MB
  off = (off + 255) & ~(size_t)255;
  _Float16* Wt = (_Float16*)(ws + off); off += (size_t)HH * FOUT * FIN * sizeof(_Float16);     // 32 KB
  off = (off + 255) & ~(size_t)255;
  float* sArr = (float*)(ws + off); off += (size_t)HH * BB * NN * sizeof(float);               // 128 KB
  float* dArr = (float*)(ws + off); off += (size_t)HH * BB * NN * sizeof(float);               // 128 KB
  float* dmax = (float*)(ws + off); off += 256;

  k_wt    <<<(HH * FOUT * FIN) / 256, 256, 0, stream>>>(W, Wt);
  k_proj  <<<(HH * BB * (NN / 16)) / 8, 256, 0, stream>>>(x, Wt, hT);
  k_scores<<<HH * BB, NN, 0, stream>>>(hT, a, sArr, dArr, dmax);
  k_attn  <<<(HH * BB * (NN / 16)) / 8, 256, 0, stream>>>(hT, sArr, dArr, dmax, out);
}
